// ResNetWithGSP_74809740361981
// MI455X (gfx1250) — compile-verified, matrix operand order checked by reading, not run on hardware
//
#include <hip/hip_runtime.h>
#include <hip/hip_bf16.h>

typedef __attribute__((ext_vector_type(16))) _Float16 v16h;
typedef __attribute__((ext_vector_type(8)))  float    v8f;
typedef __attribute__((ext_vector_type(4)))  unsigned v4u;

union AFrag { v16h v; _Float16 h[16]; unsigned u[8]; v4u q[2]; };
union CFrag { v8f v; float f[8]; };

// ---------------------------------------------------------------------------
// BN folding: scale = g * rsqrt(v+eps); shift = b - m*scale
// ---------------------------------------------------------------------------
__global__ void bn_prep_kernel(const float* __restrict__ g, const float* __restrict__ b,
                               const float* __restrict__ m, const float* __restrict__ v,
                               float* __restrict__ scale, float* __restrict__ shift, int n) {
  int i = blockIdx.x * blockDim.x + threadIdx.x;
  if (i < n) {
    float s = g[i] * rsqrtf(v[i] + 1e-5f);
    scale[i] = s;
    shift[i] = b[i] - m[i] * s;
  }
}

__global__ void bias_prep_kernel(const float* __restrict__ bias,
                                 float* __restrict__ scale, float* __restrict__ shift, int n) {
  int i = blockIdx.x * blockDim.x + threadIdx.x;
  if (i < n) { scale[i] = 1.0f; shift[i] = bias[i]; }
}

// ---------------------------------------------------------------------------
// Weight repack: fp32 OIHW -> f16 fragment-packed
// layout: [(kh*KW+kw)][coutTile][cinChunk][lane(32)][elem(16)]
// elem e -> v=e/2, half=e&1 ; K = ((v<4)?2v:2v+8) + 8*(lane>>4) + half ; N = ct*16 + (lane&15)
// ---------------------------------------------------------------------------
__global__ void repack_w_kernel(const float* __restrict__ w, _Float16* __restrict__ out,
                                int Cout, int Cin, int KH, int KW) {
  int total = KH * KW * Cout * Cin;
  int t = blockIdx.x * blockDim.x + threadIdx.x;
  if (t >= total) return;
  int e    = t & 15;
  int lane = (t >> 4) & 31;
  int c    = t >> 9;
  int nCC = Cin >> 5, nCT = Cout >> 4;
  int cc = c % nCC; c /= nCC;
  int ct = c % nCT; c /= nCT;
  int kw = c % KW;
  int kh = c / KW;
  int n  = ct * 16 + (lane & 15);
  int g  = lane >> 4;
  int vv = e >> 1, hf = e & 1;
  int K  = ((vv < 4) ? (2 * vv) : (2 * vv + 8)) + 8 * g + hf;
  int cin = cc * 32 + K;
  out[t] = (_Float16)w[(((size_t)n * Cin + cin) * KH + kh) * KW + kw];
}

// ---------------------------------------------------------------------------
// Stem: conv3x3 1->32 + BN + ReLU, input x[B=8,T=2048,F=80] viewed as image
// (H=feat=80, W=T=2048), output f16 NHWC [8,80,2048,32]
// ---------------------------------------------------------------------------
__global__ void stem_kernel(const float* __restrict__ x, const float* __restrict__ w,
                            const float* __restrict__ scale, const float* __restrict__ shift,
                            _Float16* __restrict__ y) {
  unsigned idx = blockIdx.x * blockDim.x + threadIdx.x;
  int c  = (int)(idx & 31);
  unsigned t = idx >> 5;
  int wc = (int)(t & 2047); t >>= 11;
  int h  = (int)(t % 80u);
  int b  = (int)(t / 80u);
  float acc = 0.f;
#pragma unroll
  for (int kh = 0; kh < 3; ++kh) {
    int hi = h - 1 + kh;
    if ((unsigned)hi >= 80u) continue;
#pragma unroll
    for (int kw = 0; kw < 3; ++kw) {
      int wi = wc - 1 + kw;
      if ((unsigned)wi >= 2048u) continue;
      acc += w[c * 9 + kh * 3 + kw] * x[((size_t)b * 2048 + wi) * 80 + hi];
    }
  }
  float val = acc * scale[c] + shift[c];
  val = val > 0.f ? val : 0.f;
  y[idx] = (_Float16)val;
}

// ---------------------------------------------------------------------------
// Inner K loop of the implicit-GEMM conv. CHECKED=false is the interior fast
// path: unconditional loads, no per-lane predication. Vertical (hi) borders
// are skipped with a uniform scalar branch in both paths (EXEC stays all-1s).
// A fragment = two contiguous 16B runs per lane (K = 8g..8g+7 and 16+8g..23+8g)
// -> two global_load_b128 per fragment.
// ---------------------------------------------------------------------------
template <bool CHECKED, int NCC, int NT, int MT, int KH, int KW>
__device__ __forceinline__
void conv_main(CFrag (&acc)[MT][NT], const _Float16* __restrict__ x,
               const _Float16* __restrict__ wp,
               int Hin, int Win, int b, int ho, int woT, int m, int g,
               int ct0, int nCT, int stride, int pad) {
  const int Cin = NCC * 32;
  const v4u zq = {0u, 0u, 0u, 0u};
#pragma unroll
  for (int kh = 0; kh < KH; ++kh) {
    const int hi = ho * stride - pad + kh;
    if ((unsigned)hi >= (unsigned)Hin) continue;  // uniform (ho is per-block)
#pragma unroll
    for (int kw = 0; kw < KW; ++kw) {
      bool valid[MT];
      const _Float16* xrow[MT];
#pragma unroll
      for (int mt = 0; mt < MT; ++mt) {
        const int wo_m = (woT + mt) * 16 + m;
        const int wi = wo_m * stride - pad + kw;
        valid[mt] = (unsigned)wi < (unsigned)Win;
        xrow[mt] = x + (size_t)(unsigned)(((b * Hin + hi) * Win + wi) * Cin) + 8 * g;
      }
      const _Float16* wtap = wp + (unsigned)((kh * KW + kw) * nCT) * 512u + m * 16 + g * 256;
      // note: lane*16 == (g*16 + m)*16 = g*256 + m*16
#pragma unroll
      for (int cc = 0; cc < NCC; ++cc) {
        AFrag a[MT];
#pragma unroll
        for (int mt = 0; mt < MT; ++mt) {
          const _Float16* pa = xrow[mt] + cc * 32;
          if (CHECKED) {
            a[mt].q[0] = valid[mt] ? *(const v4u*)(pa) : zq;
            a[mt].q[1] = valid[mt] ? *(const v4u*)(pa + 16) : zq;
          } else {
            a[mt].q[0] = *(const v4u*)(pa);
            a[mt].q[1] = *(const v4u*)(pa + 16);
          }
        }
#pragma unroll
        for (int j = 0; j < NT; ++j) {
          AFrag bf;
          bf.v = *(const v16h*)(wtap + (unsigned)((ct0 + j) * NCC + cc) * 512u);
#pragma unroll
          for (int mt = 0; mt < MT; ++mt) {
            acc[mt][j].v = __builtin_amdgcn_wmma_f32_16x16x32_f16(
                false, a[mt].v, false, bf.v, (short)0, acc[mt][j].v, false, false);
          }
        }
      }
    }
  }
}

// ---------------------------------------------------------------------------
// Implicit-GEMM conv via WMMA f32_16x16x32_f16, fused BN (+res)(+relu).
// Templates: NCC=Cin/32, NT cout-tiles/wave, MT m-tiles/wave, NCTG cout groups.
// Grid: x = Wout/(16*MT*8), y = Hout, z = B*NCTG. No integer division anywhere.
// Each wave: 16*MT (M=wo) x 16*NT (N=cout) output patch, MT*NT accumulators;
// A fragment reused NT times, B fragment reused MT times.
// ---------------------------------------------------------------------------
template <int NCC, int NT, int MT, int NCTG, int KH, int KW>
__global__ __launch_bounds__(256)
void conv_wmma_t(const _Float16* __restrict__ x, const _Float16* __restrict__ wp,
                 const float* __restrict__ scale, const float* __restrict__ shift,
                 const _Float16* __restrict__ res, _Float16* __restrict__ y,
                 float* __restrict__ y32,
                 int Hin, int Win, int Hout, int Wout, int Cout,
                 int stride, int pad, int relu) {
  const int lane = threadIdx.x & 31;
  const int wave = threadIdx.x >> 5;
  const int m = lane & 15, g = lane >> 4;
  const int b   = (int)blockIdx.z / NCTG;       // NCTG is a power-of-two constant
  const int ctg = (int)blockIdx.z % NCTG;
  const int ho  = (int)blockIdx.y;
  const int woT = ((int)blockIdx.x * 8 + wave) * MT;  // wo tile index (16 cols each)
  const int ct0 = ctg * NT;
  const int nCT = Cout >> 4;

  CFrag acc[MT][NT];
#pragma unroll
  for (int mt = 0; mt < MT; ++mt)
#pragma unroll
    for (int j = 0; j < NT; ++j)
#pragma unroll
      for (int i = 0; i < 8; ++i) acc[mt][j].f[i] = 0.f;

  // Horizontal borders exist only in the first/last wo-block; 1x1 convs are pad-0.
  const bool interior =
      (KH == 1) || ((blockIdx.x != 0) && (blockIdx.x + 1 != gridDim.x));
  if (interior)
    conv_main<false, NCC, NT, MT, KH, KW>(acc, x, wp, Hin, Win, b, ho, woT, m, g,
                                          ct0, nCT, stride, pad);
  else
    conv_main<true, NCC, NT, MT, KH, KW>(acc, x, wp, Hin, Win, b, ho, woT, m, g,
                                         ct0, nCT, stride, pad);

  const int orow = (b * Hout + ho) * Wout;
#pragma unroll
  for (int j = 0; j < NT; ++j) {
    const int n = (ct0 + j) * 16 + m;
    const float sc = scale[n], sh = shift[n];
#pragma unroll
    for (int mt = 0; mt < MT; ++mt) {
#pragma unroll
      for (int i = 0; i < 8; ++i) {
        const int wo_out = (woT + mt) * 16 + (i + 8 * g);
        const size_t oidx = (size_t)(unsigned)((orow + wo_out) * Cout + n);
        float v = acc[mt][j].f[i] * sc + sh;
        if (res) v += (float)res[oidx];
        if (relu) v = v > 0.f ? v : 0.f;
        if (y32) y32[oidx] = v;
        else     y[oidx] = (_Float16)v;
      }
    }
  }
}

// ---------------------------------------------------------------------------
// Time-aligned stat pooling (constants baked): feat f16 NHWC [8,10,256,256] ->
// pooled f16 [2048, 512] = [mu(256) | sigma(256)], unbiased std over n=160.
// ---------------------------------------------------------------------------
__global__ void pool_kernel(const _Float16* __restrict__ f, _Float16* __restrict__ pooled) {
  const int T = 256, C = 256, H = 10;
  unsigned idx = blockIdx.x * blockDim.x + threadIdx.x;  // over 8*256*256
  int c = (int)(idx & 255u);
  int t = (int)((idx >> 8) & 255u);
  int b = (int)(idx >> 16);
  int start = t - 8;
  if (start < 0) start = 0;
  if (start > T - 16) start = T - 16;
  float s = 0.f, q = 0.f;
  for (int l = 0; l < 16; ++l) {
    int tt = start + l;
#pragma unroll
    for (int h = 0; h < H; ++h) {
      float v = (float)f[(size_t)(unsigned)((((b * H + h) * T) + tt) * C + c)];
      s += v; q += v * v;
    }
  }
  const float n = (float)(H * 16);
  float mu = s / n;
  float var = fmaxf(q - s * mu, 0.f) / (n - 1.f);
  float sg = sqrtf(var);
  size_t base = (size_t)(unsigned)((b * T + t) * (2 * C));
  pooled[base + c]     = (_Float16)mu;
  pooled[base + C + c] = (_Float16)sg;
}

// ---------------------------------------------------------------------------
// Host-side conv dispatch over template instantiations
// ---------------------------------------------------------------------------
static void launch_conv(hipStream_t stream, const _Float16* x, const _Float16* wp,
                        const float* a, const _Float16* res, _Float16* y, float* y32,
                        int B, int Hin, int Win, int Cin, int Hout, int Wout, int Cout,
                        int KH, int stride, int pad, int relu) {
  const int NT = (Cout == 32) ? 2 : 4;
  const int NCTG = Cout / (16 * NT);  // 1,1,2,4
  dim3 grid((unsigned)(Wout / 256), (unsigned)Hout, (unsigned)(B * NCTG));
  dim3 blk(256);
  const float* sc = a;
  const float* sh = a + 512;
#define CONV_CALL(NCCv, NTv, NCTGv, KHv, KWv)                                        \
  conv_wmma_t<NCCv, NTv, 2, NCTGv, KHv, KWv><<<grid, blk, 0, stream>>>(              \
      x, wp, sc, sh, res, y, y32, Hin, Win, Hout, Wout, Cout, stride, pad, relu)
  if (KH == 3) {
    if (Cin == 32 && Cout == 32)        CONV_CALL(1, 2, 1, 3, 3);
    else if (Cin == 32 && Cout == 64)   CONV_CALL(1, 4, 1, 3, 3);
    else if (Cin == 64 && Cout == 64)   CONV_CALL(2, 4, 1, 3, 3);
    else if (Cin == 64 && Cout == 128)  CONV_CALL(2, 4, 2, 3, 3);
    else if (Cin == 128 && Cout == 128) CONV_CALL(4, 4, 2, 3, 3);
    else if (Cin == 128 && Cout == 256) CONV_CALL(4, 4, 4, 3, 3);
    else                                CONV_CALL(8, 4, 4, 3, 3);
  } else {
    if (Cin == 32)                      CONV_CALL(1, 4, 1, 1, 1);
    else if (Cin == 64)                 CONV_CALL(2, 4, 2, 1, 1);
    else if (Cin == 128)                CONV_CALL(4, 4, 4, 1, 1);
    else                                CONV_CALL(16, 4, 4, 1, 1);
  }
#undef CONV_CALL
}

// ---------------------------------------------------------------------------
// Host side
// ---------------------------------------------------------------------------
extern "C" void kernel_launch(void* const* d_in, const int* in_sizes, int n_in,
                              void* d_out, int out_size, void* d_ws, size_t ws_size,
                              hipStream_t stream) {
  (void)in_sizes; (void)n_in; (void)out_size; (void)ws_size;
  const float* x_in = (const float*)d_in[0];
  auto F = [&](int i) { return (const float*)d_in[i]; };

  // ---- workspace carve ----
  size_t off = 0;
  auto carve = [&](size_t bytes) -> void* {
    void* p = (char*)d_ws + off;
    off += (bytes + 255) & ~(size_t)255;
    return p;
  };
  const size_t BUF_ELEMS = (size_t)8 * 80 * 2048 * 32;  // largest activation
  _Float16* bufX  = (_Float16*)carve(BUF_ELEMS * 2);
  _Float16* bufT  = (_Float16*)carve(BUF_ELEMS * 2);
  _Float16* bufS  = (_Float16*)carve(BUF_ELEMS * 2);
  _Float16* pooled = (_Float16*)carve((size_t)8 * 256 * 512 * 2);
  float*    aff   = (float*)carve((size_t)21 * 1024 * 4);   // 21 slots x (512 scale + 512 shift)
  _Float16* wpack = (_Float16*)carve((size_t)3000000 * 2);

  int aslot = 0;
  auto bn_prep = [&](int bidx, int n) -> float* {  // leaves at bidx: b,g,m,v
    float* base = aff + (size_t)(aslot++) * 1024;
    bn_prep_kernel<<<(n + 255) / 256, 256, 0, stream>>>(
        F(bidx + 1), F(bidx), F(bidx + 2), F(bidx + 3), base, base + 512, n);
    return base;
  };
  size_t woff = 0;
  auto pack = [&](int widx, int Cout, int Cin, int KH, int KW) -> _Float16* {
    _Float16* dst = wpack + woff;
    int total = KH * KW * Cout * Cin;
    repack_w_kernel<<<(total + 255) / 256, 256, 0, stream>>>(F(widx), dst, Cout, Cin, KH, KW);
    woff += (size_t)total;
    return dst;
  };

  // ---- block descriptors (jax pytree leaf indices) ----
  struct Blk { int bn1, bn2, c1, c2, dbn, dcv, cin, cout, hin, win, stride; };
  const Blk blks[8] = {
      {6, 10, 14, 15, -1, -1, 32, 32, 80, 2048, 1},
      {17, 21, 25, 26, -1, -1, 32, 32, 80, 2048, 1},
      {28, 32, 36, 37, 38, 42, 32, 64, 80, 2048, 2},
      {44, 48, 52, 53, -1, -1, 64, 64, 40, 1024, 1},
      {55, 59, 63, 64, 65, 69, 64, 128, 40, 1024, 2},
      {71, 75, 79, 80, -1, -1, 128, 128, 20, 512, 1},
      {82, 86, 90, 91, 92, 96, 128, 256, 20, 512, 2},
      {98, 102, 106, 107, -1, -1, 256, 256, 10, 256, 1},
  };

  // ---- setup: BN folding + weight repack (all async on stream) ----
  float* aff_stem = bn_prep(1, 32);
  float* aff_c1[8]; float* aff_c2[8]; float* aff_dn[8];
  _Float16* wp_c1[8]; _Float16* wp_c2[8]; _Float16* wp_dn[8];
  for (int i = 0; i < 8; ++i) {
    const Blk& k = blks[i];
    aff_c1[i] = bn_prep(k.bn1, k.cout);
    aff_c2[i] = bn_prep(k.bn2, k.cout);
    wp_c1[i] = pack(k.c1, k.cout, k.cin, 3, 3);
    wp_c2[i] = pack(k.c2, k.cout, k.cout, 3, 3);
    if (k.dcv >= 0) {
      aff_dn[i] = bn_prep(k.dbn, k.cout);
      wp_dn[i] = pack(k.dcv, k.cout, k.cin, 1, 1);
    } else { aff_dn[i] = nullptr; wp_dn[i] = nullptr; }
  }
  // projection: scale=1, shift=proj_b
  float* aff_proj = aff + (size_t)(aslot++) * 1024;
  bias_prep_kernel<<<1, 256, 0, stream>>>(F(109), aff_proj, aff_proj + 512, 256);
  _Float16* wp_proj = pack(110, 256, 512, 1, 1);

  // ---- stem ----
  stem_kernel<<<(unsigned)(BUF_ELEMS / 256), 256, 0, stream>>>(
      x_in, F(5), aff_stem, aff_stem + 512, bufX);

  // ---- residual blocks ----
  _Float16* cur = bufX; _Float16* t = bufT; _Float16* s = bufS;
  for (int i = 0; i < 8; ++i) {
    const Blk& k = blks[i];
    int Hout = k.hin / k.stride, Wout = k.win / k.stride;
    // conv1 + bn1 + relu
    launch_conv(stream, cur, wp_c1[i], aff_c1[i], nullptr, t, nullptr,
                8, k.hin, k.win, k.cin, Hout, Wout, k.cout, 3, k.stride, 1, 1);
    if (k.dcv >= 0) {
      // downsample: 1x1 stride conv + bn (no relu) -> s
      launch_conv(stream, cur, wp_dn[i], aff_dn[i], nullptr, s, nullptr,
                  8, k.hin, k.win, k.cin, Hout, Wout, k.cout, 1, k.stride, 0, 0);
      // conv2 + bn2 + add(s) + relu -> cur (old contents dead)
      launch_conv(stream, t, wp_c2[i], aff_c2[i], s, cur, nullptr,
                  8, Hout, Wout, k.cout, Hout, Wout, k.cout, 3, 1, 1, 1);
      // cur now holds block output
    } else {
      // conv2 + bn2 + add(cur) + relu -> s
      launch_conv(stream, t, wp_c2[i], aff_c2[i], cur, s, nullptr,
                  8, Hout, Wout, k.cout, Hout, Wout, k.cout, 3, 1, 1, 1);
      _Float16* tmp = cur; cur = s; s = tmp;
    }
  }

  // ---- stat pooling: cur = [8,10,256,256] f16 NHWC -> pooled [2048, 512] ----
  pool_kernel<<<(unsigned)((8 * 256 * 256) / 256), 256, 0, stream>>>(cur, pooled);

  // ---- projection GEMM (as 1x1 conv): M=2048, K=512, N=256 -> d_out fp32 ----
  launch_conv(stream, pooled, wp_proj, aff_proj, nullptr, nullptr, (float*)d_out,
              1, 1, 2048, 512, 1, 2048, 256, 1, 1, 0, 0);
}
